// APPNPModel_89361089560588
// MI455X (gfx1250) — matrix operand order repaired and checked by reading, not compiled, not run on hardware
//
#include <hip/hip_runtime.h>

typedef __attribute__((ext_vector_type(16))) __bf16 v16bf;
typedef __attribute__((ext_vector_type(8)))  float  v8f;

#ifndef APPNP_ALPHA
#define APPNP_ALPHA 0.1f
#endif
#define APPNP_K 50

// ---------- helpers ----------
__device__ __forceinline__ unsigned short f2bf(float f) {
  unsigned int u = __float_as_uint(f);
  u += 0x7FFFu + ((u >> 16) & 1u);   // round-to-nearest-even
  return (unsigned short)(u >> 16);
}

// ---------- precision conversion ----------
__global__ void f32_to_bf16_kernel(const float* __restrict__ in,
                                   unsigned short* __restrict__ out, int n) {
  int i = blockIdx.x * blockDim.x + threadIdx.x;
  if (i < n) out[i] = f2bf(in[i]);
}

// W [K, Nout] row-major -> Wt [Nout, K] bf16 (so B-fragment loads are contiguous in K)
__global__ void transpose_to_bf16_kernel(const float* __restrict__ W,
                                         unsigned short* __restrict__ Wt,
                                         int K, int Nout) {
  int i = blockIdx.x * blockDim.x + threadIdx.x;
  if (i < K * Nout) {
    int k = i / Nout, n = i % Nout;
    Wt[n * K + k] = f2bf(W[i]);
  }
}

// ---------- WMMA GEMM: C = relu(A[M,K] @ B[K,Nout] + bias), A bf16, Bt = B^T bf16 ----------
// One wave32 computes one 16x16 output tile with v_wmma_f32_16x16x32_bf16.
template <bool OUT_BF16>
__global__ void gemm_relu_wmma(const unsigned short* __restrict__ A,
                               const unsigned short* __restrict__ Bt,
                               const float* __restrict__ bias,
                               void* __restrict__ Cout,
                               int M, int K, int Nout) {
  const int lane   = threadIdx.x & 31;
  const int wave   = threadIdx.x >> 5;
  const int tilesN = Nout >> 4;
  const int gw     = blockIdx.x * (blockDim.x >> 5) + wave;
  const int total  = (M >> 4) * tilesN;
  if (gw >= total) return;                    // wave-uniform exit, EXEC stays all-1s
  const int tileM = gw / tilesN;
  const int tileN = gw % tilesN;
  const int half  = lane >> 4;                // ISA 16-bit A/B lane-group select
  const int l16   = lane & 15;

  const unsigned short* arow = A  + (long)(tileM * 16 + l16) * K;
  const unsigned short* brow = Bt + (long)(tileN * 16 + l16) * K;

  v8f c = {};
  for (int kb = 0; kb < K; kb += 32) {
    union { v16bf v; unsigned int u[8]; } a, b;
#pragma unroll
    for (int e2 = 0; e2 < 8; ++e2) {
      // A 16x32 bf16: lanes 0-15 hold K=0..7,16..23 ; lanes 16-31 hold K=8..15,24..31
      int kkA = (e2 < 4) ? (2 * e2 + 8 * half) : (8 + 2 * e2 + 8 * half);
      a.u[e2] = *(const unsigned int*)(arow + kb + kkA);
      // B 32x16 bf16: same K order as A (lanes 0-15 K=0..7,16..23 ; lanes 16-31 K=8..15,24..31)
      b.u[e2] = *(const unsigned int*)(brow + kb + kkA);
    }
    c = __builtin_amdgcn_wmma_f32_16x16x32_bf16(false, a.v, false, b.v,
                                                (short)0, c, false, false);
  }

  const int col = tileN * 16 + l16;
  const float bv = bias[col];
#pragma unroll
  for (int r = 0; r < 8; ++r) {
    // C/D layout: VGPR r, lanes 0-15 -> M=r ; lanes 16-31 -> M=8+r ; N=lane&15
    int row = tileM * 16 + r + half * 8;
    float v = c[r] + bv;
    v = v > 0.0f ? v : 0.0f;
    if (OUT_BF16)
      ((unsigned short*)Cout)[(long)row * Nout + col] = f2bf(v);
    else
      ((float*)Cout)[(long)row * Nout + col] = v;
  }
}

// ---------- degree / gcn_norm ----------
__global__ void deg_init_kernel(float* __restrict__ deg, int n) {
  int i = blockIdx.x * blockDim.x + threadIdx.x;
  if (i < n) deg[i] = 1.0f;                      // self-loop
}

__global__ void deg_scatter_kernel(const int* __restrict__ col,
                                   float* __restrict__ deg, int E) {
  int e = blockIdx.x * blockDim.x + threadIdx.x;
  if (e < E)
    (void)__hip_atomic_fetch_add(&deg[col[e]], 1.0f,
                                 __ATOMIC_RELAXED, __HIP_MEMORY_SCOPE_AGENT);
}

__global__ void deg_finalize_kernel(float* __restrict__ dinv, int n) {
  int i = blockIdx.x * blockDim.x + threadIdx.x;
  if (i < n) dinv[i] = rsqrtf(dinv[i]);          // deg >= 1 always
}

// ---------- APPNP propagation (feature dim = 32, one lane per feature) ----------
__global__ void prop_init_kernel(const float* __restrict__ hc,
                                 const float* __restrict__ dinv,
                                 float* __restrict__ acc, int n32) {
  int i = blockIdx.x * blockDim.x + threadIdx.x;
  if (i < n32) {
    float d = dinv[i >> 5];
    acc[i] = d * d * hc[i];                      // self-loop contribution
  }
}

__global__ void prop_scatter_kernel(const int* __restrict__ row,
                                    const int* __restrict__ col,
                                    const float* __restrict__ dinv,
                                    const float* __restrict__ hc,
                                    float* __restrict__ acc, int E) {
  int e = blockIdx.x * (blockDim.x >> 5) + (threadIdx.x >> 5);  // wave per edge
  int f = threadIdx.x & 31;                                     // lane per feature
  if (e < E) {
    int r = row[e], c = col[e];
    float w = dinv[r] * dinv[c];
    float v = w * hc[r * 32 + f];
    (void)__hip_atomic_fetch_add(&acc[c * 32 + f], v,
                                 __ATOMIC_RELAXED, __HIP_MEMORY_SCOPE_AGENT);
  }
}

__global__ void prop_update_kernel(const float* __restrict__ acc,
                                   const float* __restrict__ h,
                                   float* __restrict__ out, int n32) {
  int i = blockIdx.x * blockDim.x + threadIdx.x;
  if (i < n32) out[i] = (1.0f - APPNP_ALPHA) * acc[i] + APPNP_ALPHA * h[i];
}

// ---------- host launch ----------
extern "C" void kernel_launch(void* const* d_in, const int* in_sizes, int n_in,
                              void* d_out, int out_size, void* d_ws, size_t ws_size,
                              hipStream_t stream) {
  const float* x  = (const float*)d_in[0];   // [N,512]
  const int*   ei = (const int*)d_in[1];     // [2,E]
  const float* W1 = (const float*)d_in[2];   // [512,256]
  const float* b1 = (const float*)d_in[3];   // [256]
  const float* W2 = (const float*)d_in[4];   // [256,32]
  const float* b2 = (const float*)d_in[5];   // [32]

  const int N = in_sizes[0] / 512;
  const int E = in_sizes[1] / 2;
  const int F1 = 512, F2 = 256, F3 = 32;
  const int* row = ei;
  const int* col = ei + E;

  // workspace carve (256B aligned)
  char* wsp = (char*)d_ws;
  auto carve = [&](size_t bytes) {
    char* p = wsp;
    wsp += (bytes + 255) & ~(size_t)255;
    return p;
  };
  unsigned short* xb  = (unsigned short*)carve((size_t)N * F1 * 2);  // x bf16
  unsigned short* w1t = (unsigned short*)carve((size_t)F2 * F1 * 2); // W1^T bf16
  unsigned short* h1b = (unsigned short*)carve((size_t)N * F2 * 2);  // h1 bf16
  unsigned short* w2t = (unsigned short*)carve((size_t)F3 * F2 * 2); // W2^T bf16
  float* h    = (float*)carve((size_t)N * F3 * 4);                   // teleport base
  float* dinv = (float*)carve((size_t)N * 4);
  float* acc  = (float*)carve((size_t)N * F3 * 4);
  float* hcA  = (float*)carve((size_t)N * F3 * 4);
  float* hcB  = (float*)carve((size_t)N * F3 * 4);

  const int T = 256;
  auto cdiv = [](long a, long b) { return (int)((a + b - 1) / b); };

  // 1) convert inputs to bf16
  f32_to_bf16_kernel<<<cdiv((long)N * F1, T), T, 0, stream>>>(x, xb, N * F1);
  transpose_to_bf16_kernel<<<cdiv((long)F1 * F2, T), T, 0, stream>>>(W1, w1t, F1, F2);
  transpose_to_bf16_kernel<<<cdiv((long)F2 * F3, T), T, 0, stream>>>(W2, w2t, F2, F3);

  // 2) MLP encoder with WMMA (bias+ReLU fused)
  {
    int waves = (N / 16) * (F2 / 16);
    gemm_relu_wmma<true><<<cdiv(waves, 8), T, 0, stream>>>(xb, w1t, b1, h1b, N, F1, F2);
  }
  {
    int waves = (N / 16) * (F3 / 16);
    gemm_relu_wmma<false><<<cdiv(waves, 8), T, 0, stream>>>(h1b, w2t, b2, h, N, F2, F3);
  }

  // 3) gcn_norm degrees -> dinv
  deg_init_kernel<<<cdiv(N, T), T, 0, stream>>>(dinv, N);
  deg_scatter_kernel<<<cdiv(E, T), T, 0, stream>>>(col, dinv, E);
  deg_finalize_kernel<<<cdiv(N, T), T, 0, stream>>>(dinv, N);

  // 4) K-step APPNP propagation (ping-pong; final step writes d_out)
  const int n32 = N * F3;
  const float* hc_cur = h;
  float* pp[2] = {hcA, hcB};
  for (int k = 0; k < APPNP_K; ++k) {
    float* hc_nxt = (k == APPNP_K - 1) ? (float*)d_out : pp[k & 1];
    prop_init_kernel<<<cdiv(n32, T), T, 0, stream>>>(hc_cur, dinv, acc, n32);
    prop_scatter_kernel<<<cdiv(E, T / 32), T, 0, stream>>>(row, col, dinv, hc_cur, acc, E);
    prop_update_kernel<<<cdiv(n32, T), T, 0, stream>>>(acc, h, hc_nxt, n32);
    hc_cur = hc_nxt;
  }
}
